// RNN_27616639714108
// MI455X (gfx1250) — matrix operand order repaired and checked by reading, not compiled, not run on hardware
//
#include <hip/hip_runtime.h>
#include <hip/hip_bf16.h>

typedef __bf16 bf16_t;
typedef __attribute__((ext_vector_type(16))) __bf16 v16bf;
typedef __attribute__((ext_vector_type(8)))  __bf16 v8bf;
typedef __attribute__((ext_vector_type(8)))  float  v8f;

#define TT 16384
#define II 512
#define HH 1024
#define CC 512
#define GG 4096           // 4*HH
#define TCHUNK 2048
#define NCHUNK 8          // TT / TCHUNK
#define LSTM_WG 64
#define LSTM_THREADS 512
#define UPW 16            // hidden units per workgroup (HH / LSTM_WG)
#define GROWS 64          // gate rows per workgroup = 4*UPW
#define WSTRIDE (HH + 1)  // LDS padding to avoid bank conflicts

static __device__ __forceinline__ bf16_t to_bf16(float f) {
  unsigned u = __builtin_bit_cast(unsigned, f);
  unsigned r = u + 0x7FFFu + ((u >> 16) & 1u);   // round-to-nearest-even
  unsigned short s = (unsigned short)(r >> 16);
  return __builtin_bit_cast(bf16_t, s);
}

static __device__ __forceinline__ float sigmoidf_(float x) {
  return 1.f / (1.f + __expf(-x));
}

// ---------------- elementwise prep ----------------
__global__ void cvt_f32_bf16(const float* __restrict__ in, bf16_t* __restrict__ out, size_t n) {
  size_t i = (size_t)blockIdx.x * blockDim.x + threadIdx.x;
  size_t stride = (size_t)gridDim.x * blockDim.x;
  for (; i < n; i += stride) out[i] = to_bf16(in[i]);
}

__global__ void bias_sum(const float* __restrict__ a, const float* __restrict__ b,
                         float* __restrict__ out, int n) {
  int i = blockIdx.x * blockDim.x + threadIdx.x;
  if (i < n) out[i] = a[i] + b[i];
}

// ---------------- bf16 WMMA GEMM:  Y[M,N] = A[M,K] * B[N,K]^T + bias[N] ----------------
// One wave computes one 16x16 tile; K-loop in steps of 32 via v_wmma_f32_16x16x32_bf16.
// A and B are both row-major with K contiguous, matching the ISA fragment layouts:
//   A 16x32 bf16: lane L holds M=L%16, K = (L>=16?8:0)+{0..7} and +16..+23  (two 16B loads)
//   B 32x16 bf16: lane L holds N=L%16, K = (L>=16?8:0)+{0..7} and +16..+23  (two 16B loads)
//   C/D 16x16 f32: vgpr i -> M = i + (L>=16?8:0), N = L%16
__global__ __launch_bounds__(256) void gemm_bf16_wmma(
    const bf16_t* __restrict__ A, const bf16_t* __restrict__ B,
    const float* __restrict__ bias, float* __restrict__ Y,
    int M, int N, int K, int arow0) {
  int ntn = N >> 4;
  int tiles = (M >> 4) * ntn;
  int tile = blockIdx.x * 8 + threadIdx.y;
  if (tile >= tiles) return;                 // wave-uniform exit (EXEC stays all-ones)
  int m0 = (tile / ntn) << 4;
  int n0 = (tile % ntn) << 4;
  int lane = threadIdx.x;
  int l15  = lane & 15;
  int hi   = lane >> 4;                      // 0 or 1

  const bf16_t* arow = A + (size_t)(arow0 + m0 + l15) * K + (hi << 3);
  const bf16_t* brow = B + (size_t)(n0 + l15) * K + (hi << 3);

  v8f acc = {};
  for (int k = 0; k < K; k += 32) {
    v8bf alo = *(const v8bf*)(arow + k);
    v8bf ahi = *(const v8bf*)(arow + k + 16);
    v16bf av = __builtin_shufflevector(alo, ahi,
        0,1,2,3,4,5,6,7,8,9,10,11,12,13,14,15);
    v8bf blo = *(const v8bf*)(brow + k);
    v8bf bhi = *(const v8bf*)(brow + k + 16);
    v16bf bv = __builtin_shufflevector(blo, bhi,
        0,1,2,3,4,5,6,7,8,9,10,11,12,13,14,15);
    acc = __builtin_amdgcn_wmma_f32_16x16x32_bf16(
        false, av, false, bv, (short)0, acc, false, false);
  }

  float bval = bias[n0 + l15];
  int mbase = m0 + (hi << 3);
  size_t col = (size_t)(n0 + l15);
  #pragma unroll
  for (int i = 0; i < 8; ++i)
    Y[(size_t)(mbase + i) * N + col] = acc[i] + bval;
}

// ---------------- persistent LSTM recurrence ----------------
// 64 WGs x 512 threads. WG owns 16 hidden units -> 64 gate rows of W_hh in LDS (fp32).
// Per step: gates = Xproj[t] + W_hh_slice * h ; LSTM cell update; grid barrier; reload h.
__global__ __launch_bounds__(512) void lstm_chunk(
    const float* __restrict__ Xproj,   // [TCHUNK, GG] for this chunk
    const float* __restrict__ Whh,     // [GG, HH]
    float* __restrict__ h_state,       // [HH]
    float* __restrict__ c_state,       // [HH]
    bf16_t* __restrict__ Hout,         // [TT, HH]
    unsigned* __restrict__ syncc,
    int t0) {
  extern __shared__ float smem[];
  float* Wlds = smem;                      // GROWS * WSTRIDE
  float* hbuf = Wlds + GROWS * WSTRIDE;    // HH
  float* pbuf = hbuf + HH;                 // 8 * 64 partial sums
  float* gbuf = pbuf + 8 * 64;             // 64 gate pre-activations
  float* cbuf = gbuf + 64;                 // 16 cell states

  int tid = threadIdx.x;
  int jbase = blockIdx.x * UPW;

  // stage the W_hh slice (rows: gt*HH + jbase + u for gt=0..3, u=0..15)
  for (int idx = tid; idx < GROWS * HH; idx += LSTM_THREADS) {
    int lr = idx >> 10;                    // idx / HH
    int k  = idx & (HH - 1);
    int gt = lr >> 4, u = lr & 15;
    Wlds[lr * WSTRIDE + k] = Whh[(size_t)(gt * HH + jbase + u) * HH + k];
  }
  for (int j = tid; j < HH; j += LSTM_THREADS) hbuf[j] = h_state[j];
  if (tid < UPW) cbuf[tid] = c_state[jbase + tid];
  __syncthreads();

  int lr  = tid & 63;                      // local gate row
  int seg = tid >> 6;                      // k-segment 0..7 (128 k each)
  const float* wr = Wlds + lr * WSTRIDE + (seg << 7);
  const float* hr = hbuf + (seg << 7);

  for (int tl = 0; tl < TCHUNK; ++tl) {
    float acc = 0.f;
    #pragma unroll 8
    for (int s = 0; s < 128; ++s) acc = __builtin_fmaf(wr[s], hr[s], acc);
    pbuf[(seg << 6) + lr] = acc;
    __syncthreads();

    if (tid < 64) {
      int gt = tid >> 4, u = tid & 15;
      float g = Xproj[(size_t)tl * GG + gt * HH + jbase + u];
      #pragma unroll
      for (int s = 0; s < 8; ++s) g += pbuf[(s << 6) + tid];
      gbuf[tid] = g;
    }
    __syncthreads();

    if (tid < UPW) {
      float iv = sigmoidf_(gbuf[tid]);
      float fv = sigmoidf_(gbuf[UPW + tid]);
      float gv = tanhf(gbuf[2 * UPW + tid]);
      float ov = sigmoidf_(gbuf[3 * UPW + tid]);
      float c  = fv * cbuf[tid] + iv * gv;
      cbuf[tid] = c;
      float h  = ov * tanhf(c);
      h_state[jbase + tid] = h;
      Hout[(size_t)(t0 + tl) * HH + jbase + tid] = to_bf16(h);
    }
    __threadfence();
    __syncthreads();

    // device-wide step barrier
    if (tid == 0) {
      __hip_atomic_fetch_add(syncc, 1u, __ATOMIC_RELEASE, __HIP_MEMORY_SCOPE_AGENT);
      unsigned target = (unsigned)gridDim.x * (unsigned)(t0 + tl + 1);
      while (__hip_atomic_load(syncc, __ATOMIC_ACQUIRE, __HIP_MEMORY_SCOPE_AGENT) < target)
        __builtin_amdgcn_s_sleep(2);
    }
    __syncthreads();
    for (int j = tid; j < HH; j += LSTM_THREADS) hbuf[j] = h_state[j];
    __syncthreads();
  }
  if (tid < UPW) c_state[jbase + tid] = cbuf[tid];
}

// ---------------- softmax over channels (per row) ----------------
__global__ __launch_bounds__(256) void row_softmax(float* __restrict__ Y, int N) {
  __shared__ float red[256];
  int row = blockIdx.x, tid = threadIdx.x;
  float* p = Y + (size_t)row * N;
  float m = -3.4e38f;
  for (int j = tid; j < N; j += 256) m = fmaxf(m, p[j]);
  red[tid] = m; __syncthreads();
  for (int s = 128; s > 0; s >>= 1) { if (tid < s) red[tid] = fmaxf(red[tid], red[tid + s]); __syncthreads(); }
  m = red[0]; __syncthreads();
  float sum = 0.f;
  for (int j = tid; j < N; j += 256) { float e = __expf(p[j] - m); p[j] = e; sum += e; }
  red[tid] = sum; __syncthreads();
  for (int s = 128; s > 0; s >>= 1) { if (tid < s) red[tid] += red[tid + s]; __syncthreads(); }
  float inv = 1.f / red[0];
  for (int j = tid; j < N; j += 256) p[j] *= inv;
}

// ---------------- softmax over time (per channel column) ----------------
__global__ __launch_bounds__(256) void time_softmax(float* __restrict__ Y, int T, int C) {
  __shared__ float red[8][32];
  int cx = threadIdx.x, ty = threadIdx.y;
  int c = blockIdx.x * 32 + cx;
  float m = -3.4e38f;
  for (int t = ty; t < T; t += 8) m = fmaxf(m, Y[(size_t)t * C + c]);
  red[ty][cx] = m; __syncthreads();
  if (ty == 0) { for (int s = 1; s < 8; ++s) m = fmaxf(m, red[s][cx]); red[0][cx] = m; }
  __syncthreads();
  m = red[0][cx]; __syncthreads();
  float sum = 0.f;
  for (int t = ty; t < T; t += 8) {
    size_t idx = (size_t)t * C + c;
    float e = __expf(Y[idx] - m); Y[idx] = e; sum += e;
  }
  red[ty][cx] = sum; __syncthreads();
  if (ty == 0) { for (int s = 1; s < 8; ++s) sum += red[s][cx]; red[0][cx] = sum; }
  __syncthreads();
  float inv = 1.f / red[0][cx];
  for (int t = ty; t < T; t += 8) Y[(size_t)t * C + c] *= inv;
}

// ---------------- host orchestration ----------------
extern "C" void kernel_launch(void* const* d_in, const int* in_sizes, int n_in,
                              void* d_out, int out_size, void* d_ws, size_t ws_size,
                              hipStream_t stream) {
  (void)in_sizes; (void)n_in; (void)out_size; (void)ws_size;
  const float* X   = (const float*)d_in[0];
  const float* h0  = (const float*)d_in[1];
  const float* c0  = (const float*)d_in[2];
  const float* Wih = (const float*)d_in[3];
  const float* Whh = (const float*)d_in[4];
  const float* bih = (const float*)d_in[5];
  const float* bhh = (const float*)d_in[6];
  const float* Wd  = (const float*)d_in[7];
  const float* bd  = (const float*)d_in[8];
  float* out = (float*)d_out;

  char* ws = (char*)d_ws;
  size_t off = 0;
  auto take = [&](size_t bytes) -> void* {
    void* p = ws + off;
    off = (off + bytes + 255) & ~(size_t)255;
    return p;
  };
  bf16_t*   Xb    = (bf16_t*)take((size_t)TT * II * 2);       // 16 MB
  bf16_t*   Wihb  = (bf16_t*)take((size_t)GG * II * 2);       //  4 MB
  bf16_t*   Wdb   = (bf16_t*)take((size_t)CC * HH * 2);       //  1 MB
  float*    bias  = (float*) take((size_t)GG * 4);
  float*    Xproj = (float*) take((size_t)TCHUNK * GG * 4);   // 32 MB
  bf16_t*   Houtb = (bf16_t*)take((size_t)TT * HH * 2);       // 32 MB
  float*    hst   = (float*) take((size_t)HH * 4);
  float*    cst   = (float*) take((size_t)HH * 4);
  unsigned* syncc = (unsigned*)take(256);

  hipMemsetAsync(syncc, 0, 256, stream);
  hipMemcpyAsync(hst, h0, HH * sizeof(float), hipMemcpyDeviceToDevice, stream);
  hipMemcpyAsync(cst, c0, HH * sizeof(float), hipMemcpyDeviceToDevice, stream);

  cvt_f32_bf16<<<4096, 256, 0, stream>>>(X,   Xb,   (size_t)TT * II);
  cvt_f32_bf16<<<2048, 256, 0, stream>>>(Wih, Wihb, (size_t)GG * II);
  cvt_f32_bf16<<<512,  256, 0, stream>>>(Wd,  Wdb,  (size_t)CC * HH);
  bias_sum<<<GG / 256, 256, 0, stream>>>(bih, bhh, bias, GG);

  size_t lds_bytes = (size_t)(GROWS * WSTRIDE + HH + 8 * 64 + 64 + 16) * 4;
  for (int ch = 0; ch < NCHUNK; ++ch) {
    int t0 = ch * TCHUNK;
    int tiles = (TCHUNK / 16) * (GG / 16);
    gemm_bf16_wmma<<<(tiles + 7) / 8, dim3(32, 8), 0, stream>>>(
        Xb, Wihb, bias, Xproj, TCHUNK, GG, II, t0);
    lstm_chunk<<<LSTM_WG, LSTM_THREADS, lds_bytes, stream>>>(
        Xproj, Whh, hst, cst, Houtb, syncc, t0);
  }

  {
    int tiles = (TT / 16) * (CC / 16);
    gemm_bf16_wmma<<<(tiles + 7) / 8, dim3(32, 8), 0, stream>>>(
        Houtb, Wdb, bd, out, TT, CC, HH, 0);
  }
  row_softmax<<<TT, 256, 0, stream>>>(out, CC);
  time_softmax<<<CC / 32, dim3(32, 8), 0, stream>>>(out, TT, CC);
}
